// HGSA_v9_47244640256608
// MI455X (gfx1250) — matrix operand order repaired and checked by reading, not compiled, not run on hardware
//
#include <hip/hip_runtime.h>
#include <hip/hip_bf16.h>
#include <math.h>

typedef unsigned short u16;
typedef __attribute__((ext_vector_type(16))) __bf16 v16bf;
typedef __attribute__((ext_vector_type(8)))  float  v8f;
typedef __attribute__((ext_vector_type(4)))  unsigned int u32x4;

#define BATCH 4
#define HH    256
#define WW    256
#define HWSZ  (HH*WW)
#define TOK   (BATCH*HWSZ)
#define CDIV(a,b) (((a)+(b)-1)/(b))

union Frag { v16bf v; u32x4 q[2]; u16 h[16]; };

__device__ __forceinline__ u16 f2bf(float f) {
  return __builtin_bit_cast(u16, __float2bfloat16(f));
}
__device__ __forceinline__ float gelu_f(float x){ return 0.5f*x*(1.f+erff(x*0.70710678118654752f)); }
__device__ __forceinline__ float sigm_f(float x){ return 1.f/(1.f+expf(-x)); }

// ---------------------------------------------------------------------------
// Weight prep: convert f32 weight to bf16 in [Npad, K] row-major (K-contiguous)
// transB=1: W already [N,K] (OIHW 1x1); transB=0: W is [K,N] -> transpose.
// Rows >= N are zero-filled so GEMM async staging never needs a bounds check.
// ---------------------------------------------------------------------------
__global__ void k_wprep(const float* __restrict__ W, u16* __restrict__ out,
                        int N, int K, int transB, int n)
{
  int idx = blockIdx.x * blockDim.x + threadIdx.x;
  if (idx >= n) return;
  int nn = idx / K, k = idx % K;
  float v = 0.f;
  if (nn < N) v = transB ? W[(size_t)nn * K + k] : W[(size_t)k * N + nn];
  out[idx] = f2bf(v);
}

// ---------------------------------------------------------------------------
// Fused GEMM: C[M,N] = act( A[M,K](f32->bf16 in-stage) @ Bw[bf16 Npad,K] + bias + res )
// out_mode=0: token-major store; out_mode=1: NCHW scatter into d_out.
// M multiple of 128, K multiple of 32. B tile staged via async DMA to LDS.
// ---------------------------------------------------------------------------
#define GM 128
#define GN 32
#define GK 32
#define GP 8
__global__ __launch_bounds__(256) void k_gemm_wmma(
    const float* __restrict__ A, const u16* __restrict__ Bw,
    const float* __restrict__ bias, const float* __restrict__ res,
    float* __restrict__ C, int M, int N, int K, int act,
    int out_mode, int Ctot, int c0)
{
  __shared__ __align__(16) u16 As[GM][GK+GP];  // 80B row stride: 16B-aligned frags
  __shared__ __align__(16) u16 Bs[GN][GK+GP];
  const int tid  = threadIdx.x;
  const int m0   = blockIdx.x * GM;
  const int n0   = blockIdx.y * GN;
  const int wave = tid >> 5, lane = tid & 31;
  const int lrow = lane & 15;
  const int hi   = (lane >= 16);

  v8f z8 = {0.f,0.f,0.f,0.f,0.f,0.f,0.f,0.f};
  v8f acc[2]; acc[0] = z8; acc[1] = z8;

  const int arow = tid >> 1;          // 0..127
  const int acol = (tid & 1) << 4;    // 0 or 16
  const int brow = tid >> 3;          // 0..31 (B tile row)
  const int bchk = (tid & 7) << 2;    // u16 offset within row: 0,4,..,28
  const unsigned ldsB = (unsigned)(size_t)&Bs[brow][bchk];

  for (int kb = 0; kb < K; kb += GK) {
    // ---- B tile: bf16 weights, async DMA straight into LDS (8B per lane) ----
    unsigned goff = (((unsigned)(n0 + brow) * (unsigned)K) + (unsigned)kb + (unsigned)bchk) * 2u;
    asm volatile("global_load_async_to_lds_b64 %0, %1, %2"
                 :: "v"(ldsB), "v"(goff), "s"(Bw) : "memory");
    // ---- A tile: single-pass f32 load + bf16 convert into LDS ----
    const float* ag = A + (size_t)(m0 + arow) * K + kb + acol;
    __builtin_prefetch((const void*)(ag + GK), 0, 0);   // global_prefetch_b8
    #pragma unroll
    for (int j = 0; j < 16; ++j) { As[arow][acol + j] = f2bf(ag[j]); }
    asm volatile("s_wait_asynccnt 0" ::: "memory");
    __syncthreads();

    Frag fa, fb0, fb1;
    const int ak = hi ? 16 : 0;  // A 16x32 bf16 lane layout
    fa.q[0] = *(const u32x4*)&As[wave*16 + lrow][ak];
    fa.q[1] = *(const u32x4*)&As[wave*16 + lrow][ak + 8];
    const int bk = hi ? 16 : 0;  // B 32x16 bf16 lane layout
    fb0.q[0] = *(const u32x4*)&Bs[lrow][bk];
    fb0.q[1] = *(const u32x4*)&Bs[lrow][bk + 8];
    fb1.q[0] = *(const u32x4*)&Bs[16 + lrow][bk];
    fb1.q[1] = *(const u32x4*)&Bs[16 + lrow][bk + 8];
    acc[0] = __builtin_amdgcn_wmma_f32_16x16x32_bf16(false, fa.v, false, fb0.v, (short)0, acc[0], false, false);
    acc[1] = __builtin_amdgcn_wmma_f32_16x16x32_bf16(false, fa.v, false, fb1.v, (short)0, acc[1], false, false);
    __syncthreads();
  }

  #pragma unroll
  for (int j = 0; j < 2; ++j) {
    int col = n0 + j*16 + lrow;
    if (col >= N) continue;
    float bv = bias ? bias[col] : 0.f;
    #pragma unroll
    for (int r = 0; r < 8; ++r) {
      int m = m0 + wave*16 + (hi ? 8 : 0) + r;   // C layout: lanes 16-31 -> M+8
      float v = acc[j][r] + bv;
      if (res) v += res[(size_t)m * N + col];
      if (act == 1) v = gelu_f(v);
      if (out_mode == 0) {
        C[(size_t)m * N + col] = v;
      } else {
        int b = m / HWSZ, p = m % HWSZ;
        C[((size_t)b * Ctot + c0 + col) * HWSZ + p] = v;
      }
    }
  }
}

// ---------------------------------------------------------------------------
// Attention logits: attn = softmax( rescale * (K/|K|) @ (Q/|Q|)^T ) per (b,h)
// Gram over n=HWSZ computed with bf16 WMMA (K-dim = n).
// attnw layout: [(b*heads+h)][32][32] (fixed 32 stride, only dh x dh valid)
// ---------------------------------------------------------------------------
__global__ __launch_bounds__(256) void k_attn(
    const float* __restrict__ qkv, int ld, int hd, int dh, int heads,
    const float* __restrict__ rescale, float* __restrict__ attnw)
{
  const int bh = blockIdx.x;
  const int b = bh / heads, h = bh % heads;
  const size_t t0 = (size_t)b * HWSZ;
  const int qoff = h * dh, koff = hd + h * dh;
  const int tid = threadIdx.x;
  const int wave = tid >> 5, lane = tid & 31, lrow = lane & 15;
  const int hi = (lane >= 16);
  const int DT = dh >> 4;                 // 1 (dh=16) or 2 (dh=32)

  __shared__ float s_norm[64];            // [0..31]=q sumsq, [32..63]=k sumsq
  __shared__ float s_part[8][1024];       // per-wave partial 32x32 tiles
  if (tid < 64) s_norm[tid] = 0.f;
  __syncthreads();
  { // phase 1: row L2 norms of q and k over n
    int grp = tid & 3, r = (tid >> 2) & 31, which = tid >> 7;
    if (r < dh) {
      int off = which ? koff : qoff;
      float s = 0.f;
      for (int n = grp; n < HWSZ; n += 4) {
        float x = qkv[(t0 + n) * ld + off + r];
        s += x * x;
      }
      atomicAdd(&s_norm[which * 32 + r], s);
    }
  }

  v8f z8 = {0.f,0.f,0.f,0.f,0.f,0.f,0.f,0.f};
  v8f acc[2][2];
  acc[0][0]=z8; acc[0][1]=z8; acc[1][0]=z8; acc[1][1]=z8;

  for (int n0 = wave * 32; n0 < HWSZ; n0 += 256) {
    Frag fa[2], fb[2];
    for (int dt = 0; dt < DT; ++dt) {
      int row = dt * 16 + lrow;           // K-matrix row (d index)
      #pragma unroll
      for (int j = 0; j < 16; ++j) {
        int kk = (hi ? 8 : 0) + j + ((j >= 8) ? 8 : 0);   // A lane K layout
        fa[dt].h[j] = f2bf(qkv[(t0 + n0 + kk) * ld + koff + row]);
      }
    }
    for (int et = 0; et < DT; ++et) {
      int col = et * 16 + lrow;           // Q-matrix row (e index)
      #pragma unroll
      for (int j = 0; j < 16; ++j) {
        int kk = (hi ? 8 : 0) + j + ((j >= 8) ? 8 : 0);   // B lane K layout
        fb[et].h[j] = f2bf(qkv[(t0 + n0 + kk) * ld + qoff + col]);
      }
    }
    for (int dt = 0; dt < DT; ++dt) {
      for (int et = 0; et < DT; ++et) {
        acc[dt][et] = __builtin_amdgcn_wmma_f32_16x16x32_bf16(
            false, fa[dt].v, false, fb[et].v, (short)0, acc[dt][et], false, false);
      }
    }
  }

  for (int dt = 0; dt < DT; ++dt) {
    for (int et = 0; et < DT; ++et) {
      #pragma unroll
      for (int r = 0; r < 8; ++r) {
        int drow = dt * 16 + (hi ? 8 : 0) + r;
        int ecol = et * 16 + lrow;
        s_part[wave][drow * 32 + ecol] = acc[dt][et][r];
      }
    }
  }
  __syncthreads();

  if (tid < dh) { // cross-wave reduce + normalize + softmax over e
    float kn = fmaxf(sqrtf(s_norm[32 + tid]), 1e-12f);
    float rs = rescale[h];
    float vals[32];
    float mx = -1e30f;
    for (int e = 0; e < dh; ++e) {
      float s = 0.f;
      for (int w = 0; w < 8; ++w) s += s_part[w][tid * 32 + e];
      float qn = fmaxf(sqrtf(s_norm[e]), 1e-12f);
      float v = s / (kn * qn) * rs;
      vals[e] = v;
      mx = fmaxf(mx, v);
    }
    float sum = 0.f;
    for (int e = 0; e < dh; ++e) { vals[e] = expf(vals[e] - mx); sum += vals[e]; }
    float inv = 1.f / sum;
    for (int e = 0; e < dh; ++e)
      attnw[(size_t)bh * 1024 + tid * 32 + e] = vals[e] * inv;
  }
}

// out[t,h*dh+d] = sum_e attn[h][d][e] * (v[t,e]*mf[t,e]); also emits masked v.
__global__ __launch_bounds__(256) void k_apply(
    const float* __restrict__ qkv, int ld, int hd, int dh, int heads,
    const float* __restrict__ mf, const float* __restrict__ attnw,
    float* __restrict__ vmask, float* __restrict__ outc)
{
  __shared__ float s_attn[2048];
  const int tid = threadIdx.x;
  const size_t t = (size_t)blockIdx.x * 256 + tid;
  const int b = (int)(((size_t)blockIdx.x * 256) / HWSZ);
  const int tot = heads * 1024;
  for (int i = tid; i < tot; i += 256)
    s_attn[i] = attnw[(size_t)b * heads * 1024 + i];
  __syncthreads();
  for (int h = 0; h < heads; ++h) {
    float vv[32];
    int voff = 2 * hd + h * dh;
    for (int e = 0; e < dh; ++e) {
      float x = qkv[t * ld + voff + e] * mf[t * hd + h * dh + e];
      vv[e] = x;
      vmask[t * hd + h * dh + e] = x;
    }
    for (int d = 0; d < dh; ++d) {
      float s = 0.f;
      const float* ar = &s_attn[h * 1024 + d * 32];
      for (int e = 0; e < dh; ++e) s += ar[e] * vv[e];
      outc[t * hd + h * dh + d] = s;
    }
  }
}

// depthwise conv, token-major [T,C], odd ks, zero pad; act=1 -> gelu
__global__ void k_dwconv(const float* __restrict__ in, const float* __restrict__ wgt,
                         const float* __restrict__ bias, float* __restrict__ out,
                         int C, int ks, int pad, int act, int n)
{
  int idx = blockIdx.x * blockDim.x + threadIdx.x;
  if (idx >= n) return;
  int c = idx % C;
  long t = idx / C;
  int p = (int)(t % HWSZ);
  int y = p / WW, x = p % WW;
  float s = bias ? bias[c] : 0.f;
  for (int ky = 0; ky < ks; ++ky) {
    int yy = y + ky - pad;
    if (yy < 0 || yy >= HH) continue;
    for (int kx = 0; kx < ks; ++kx) {
      int xx = x + kx - pad;
      if (xx < 0 || xx >= WW) continue;
      s += in[((size_t)t + (yy - y) * WW + (xx - x)) * C + c] * wgt[(c * ks + ky) * ks + kx];
    }
  }
  if (act == 1) s = gelu_f(s);
  out[(size_t)t * C + c] = s;
}

// 1x1 conv: NCHW input -> token-major output (for mgm pm: 3->32)
__global__ void k_pw_nchw_tok(const float* __restrict__ xin, const float* __restrict__ wgt,
                              const float* __restrict__ bias, float* __restrict__ out,
                              int Ci, int Co, int n)
{
  int idx = blockIdx.x * blockDim.x + threadIdx.x;
  if (idx >= n) return;
  int o = idx % Co;
  size_t t = (size_t)(idx / Co);
  int b = (int)(t / HWSZ), p = (int)(t % HWSZ);
  float s = bias ? bias[o] : 0.f;
  for (int i = 0; i < Ci; ++i)
    s += wgt[o * Ci + i] * xin[((size_t)b * Ci + i) * HWSZ + p];
  out[t * Co + o] = s;
}

// dense 3x3 conv (pad 1), NCHW input -> token-major output (init conv, xi conv)
__global__ void k_conv3x3(const float* __restrict__ xin, const float* __restrict__ wgt,
                          const float* __restrict__ bias, float* __restrict__ out,
                          int Ci, int Co, int n)
{
  int idx = blockIdx.x * blockDim.x + threadIdx.x;
  if (idx >= n) return;
  int o = idx % Co;
  size_t t = (size_t)(idx / Co);
  int b = (int)(t / HWSZ), p = (int)(t % HWSZ);
  int y = p / WW, x = p % WW;
  float s = bias[o];
  for (int i = 0; i < Ci; ++i) {
    for (int ky = 0; ky < 3; ++ky) {
      int yy = y + ky - 1;
      if (yy < 0 || yy >= HH) continue;
      for (int kx = 0; kx < 3; ++kx) {
        int xx = x + kx - 1;
        if (xx < 0 || xx >= WW) continue;
        s += wgt[((o * Ci + i) * 3 + ky) * 3 + kx] * xin[((size_t)b * Ci + i) * HWSZ + yy * WW + xx];
      }
    }
  }
  out[t * Co + o] = s;
}

// per-token LayerNorm over channel (token-major)
__global__ void k_ln(const float* __restrict__ in, const float* __restrict__ g,
                     const float* __restrict__ be, float* __restrict__ out, int C, int M)
{
  int t = blockIdx.x * blockDim.x + threadIdx.x;
  if (t >= M) return;
  float buf[32];
  const float* p = in + (size_t)t * C;
  float m = 0.f;
  for (int c = 0; c < C; ++c) { buf[c] = p[c]; m += buf[c]; }
  m /= C;
  float v = 0.f;
  for (int c = 0; c < C; ++c) { float d = buf[c] - m; v += d * d; }
  v /= C;
  float r = rsqrtf(v + 1e-5f);
  for (int c = 0; c < C; ++c) out[(size_t)t * C + c] = (buf[c] - m) * r * g[c] + be[c];
}

// LayerNorm over channel then scatter NCHW into d_out at channel offset c0
__global__ void k_ln_nchw(const float* __restrict__ in, const float* __restrict__ g,
                          const float* __restrict__ be, float* __restrict__ out,
                          int C, int Ctot, int c0)
{
  int t = blockIdx.x * blockDim.x + threadIdx.x;
  if (t >= TOK) return;
  float buf[4];
  const float* p = in + (size_t)t * C;
  float m = 0.f;
  for (int c = 0; c < C; ++c) { buf[c] = p[c]; m += buf[c]; }
  m /= C;
  float v = 0.f;
  for (int c = 0; c < C; ++c) { float d = buf[c] - m; v += d * d; }
  v /= C;
  float r = rsqrtf(v + 1e-5f);
  int b = t / HWSZ, pp = t % HWSZ;
  for (int c = 0; c < C; ++c)
    out[((size_t)b * Ctot + c0 + c) * HWSZ + pp] = (buf[c] - m) * r * g[c] + be[c];
}

__global__ void k_pool_nchw(const float* __restrict__ xin, float* __restrict__ out)
{
  int bc = blockIdx.x, tid = threadIdx.x;
  __shared__ float red[256];
  float s = 0.f;
  for (int p = tid; p < HWSZ; p += 256) s += xin[(size_t)bc * HWSZ + p];
  red[tid] = s; __syncthreads();
  for (int k = 128; k > 0; k >>= 1) { if (tid < k) red[tid] += red[tid + k]; __syncthreads(); }
  if (tid == 0) out[bc] = red[0] / (float)HWSZ;
}

__global__ void k_pool_tok(const float* __restrict__ in, float* __restrict__ out, int C)
{
  int bc = blockIdx.x, tid = threadIdx.x;
  int b = bc / C, c = bc % C;
  __shared__ float red[256];
  float s = 0.f;
  for (int p = tid; p < HWSZ; p += 256) s += in[((size_t)b * HWSZ + p) * C + c];
  red[tid] = s; __syncthreads();
  for (int k = 128; k > 0; k >>= 1) { if (tid < k) red[tid] += red[tid + k]; __syncthreads(); }
  if (tid == 0) out[bc] = red[0] / (float)HWSZ;
}

// GCE: ctx[b,g,c] = silu(vpool @ W1 + b1) @ W2 + b2, reshaped [B,8,32]
__global__ void k_gce(const float* __restrict__ vpool, const float* __restrict__ w1,
                      const float* __restrict__ b1, const float* __restrict__ w2,
                      const float* __restrict__ b2, float* __restrict__ ctx)
{
  __shared__ float hbuf[BATCH * 32];
  int tid = threadIdx.x;
  if (tid < BATCH * 32) {
    int b = tid >> 5, j = tid & 31;
    float s = b1[j];
    for (int i = 0; i < 3; ++i) s += vpool[b * 3 + i] * w1[i * 32 + j];
    hbuf[tid] = s * sigm_f(s);   // silu
  }
  __syncthreads();
  for (int o = tid; o < BATCH * 256; o += 256) {
    int b = o >> 8, c = o & 255;
    float s = b2[c];
    for (int j = 0; j < 32; ++j) s += hbuf[b * 32 + j] * w2[j * 256 + c];
    ctx[o] = s;
  }
}

__global__ void k_concat(const float* __restrict__ z, const float* __restrict__ ctx,
                         int g, float* __restrict__ xq, int n)
{
  int idx = blockIdx.x * blockDim.x + threadIdx.x;
  if (idx >= n) return;
  int c = idx & 63;
  size_t t = (size_t)(idx >> 6);
  int b = (int)(t / HWSZ);
  xq[idx] = (c < 32) ? z[t * 32 + c] : ctx[b * 256 + g * 32 + (c - 32)];
}

__global__ void k_combine(const float* __restrict__ m, const float* __restrict__ a,
                          float* __restrict__ out, int n)   // m*sigmoid(a)+m
{
  int i = blockIdx.x * blockDim.x + threadIdx.x;
  if (i >= n) return;
  float mm = m[i];
  out[i] = mm * sigm_f(a[i]) + mm;
}

__global__ void k_add3(float* __restrict__ d, const float* __restrict__ a,
                       const float* __restrict__ b, const float* __restrict__ c, int n)
{
  int i = blockIdx.x * blockDim.x + threadIdx.x;
  if (i >= n) return;
  d[i] = a[i] + b[i] + c[i];
}

__global__ void k_xi_gate(const float* __restrict__ g, const float* __restrict__ w1,
                          const float* __restrict__ b1, const float* __restrict__ w2,
                          const float* __restrict__ b2, float* __restrict__ gate)
{
  __shared__ float hb[16];
  int b = blockIdx.x, t = threadIdx.x;
  if (t < 16) {
    float s = b1[t];
    for (int i = 0; i < 16; ++i) s += g[b * 16 + i] * w1[t * 16 + i];
    hb[t] = s * sigm_f(s);
  }
  __syncthreads();
  if (t < 16) {
    float s = b2[t];
    for (int i = 0; i < 16; ++i) s += hb[i] * w2[t * 16 + i];
    gate[b * 16 + t] = sigm_f(s);
  }
}

__global__ void k_xi_combine(const float* __restrict__ lc, const float* __restrict__ gate,
                             const float* __restrict__ z, float* __restrict__ out, int n)
{
  int i = blockIdx.x * blockDim.x + threadIdx.x;
  if (i >= n) return;
  int c = i & 15;
  int b = (int)(((size_t)(i >> 4)) / HWSZ);
  out[i] = lc[i] * gate[b * 16 + c] + z[i];
}

__global__ void k_pw_tok(const float* __restrict__ in, const float* __restrict__ wgt,
                         const float* __restrict__ bias, float* __restrict__ out,
                         int Ci, int Co, int n)
{
  int idx = blockIdx.x * blockDim.x + threadIdx.x;
  if (idx >= n) return;
  int o = idx % Co;
  size_t t = (size_t)(idx / Co);
  float s = bias[o];
  for (int i = 0; i < Ci; ++i) s += wgt[o * Ci + i] * in[t * Ci + i];
  out[t * Co + o] = s;
}

// ---------------------------------------------------------------------------
// Host side
// ---------------------------------------------------------------------------
struct BlkW {  // leaf order = jax sorted-key flatten order within a block dict
  const float *ff1,*ff2,*ffd,*ln_b,*ln_g;
  const float *c1_b,*c1_w,*c2_b,*c2_w,*dw_b,*dw_w,*pm_b,*pm_w;
  const float *pos1,*pos2,*proj_b,*proj_w,*qkv,*rescale;
};
static_assert(sizeof(BlkW) == 19 * sizeof(void*), "BlkW layout");

static BlkW blk_at(void* const* in, int base) {
  BlkW b;
  const float** f = (const float**)&b;
  for (int i = 0; i < 19; ++i) f[i] = (const float*)in[base + i];
  return b;
}

struct WS {
  float *vpool, *ctx, *feat, *zA, *zB, *zC, *xq, *qkvb;
  float *m0, *m1, *t0b, *t1b, *mf, *attnw, *vmask, *outc, *proj, *p1, *p2, *lnb, *f1, *f2;
  float *z16, *zn16, *g16, *gate, *lc, *comb, *t3;
  u16   *wB;
};

// prep weights to bf16 [Npad,K] then run the WMMA GEMM
static void gemm(hipStream_t st, const WS& w, const float* A, const float* Wt,
                 const float* bias, const float* res, float* C,
                 int N, int K, int transB, int act, int out_mode, int Ctot, int c0)
{
  int Npad = CDIV(N, 32) * 32;
  k_wprep<<<CDIV(Npad*K,256),256,0,st>>>(Wt, w.wB, N, K, transB, Npad*K);
  dim3 g(TOK/GM, CDIV(N, 32));
  k_gemm_wmma<<<g,256,0,st>>>(A, w.wB, bias, res, C, TOK, N, K, act, out_mode, Ctot, c0);
}

static void msab_block(hipStream_t st, const WS& w, const float* xin,
                       const float* zin, float* zout, const BlkW& bw,
                       const float* ctx, int g, int heads, int dh)
{
  const int hd = 32, ld = 96;
  const float* Aq = zin; int Kq = 32;
  if (ctx) {
    k_concat<<<CDIV(TOK*64,256),256,0,st>>>(zin, ctx, g, w.xq, TOK*64);
    Aq = w.xq; Kq = 64;
  }
  gemm(st, w, Aq, bw.qkv, nullptr, nullptr, w.qkvb, 96, Kq, 0, 0, 0, 0, 0);
  // mgm: m = c1(pm(mask)); a = sigmoid(dw5(c2(m))); mf = m*a + m
  k_pw_nchw_tok<<<CDIV(TOK*32,256),256,0,st>>>(xin, bw.pm_w, bw.pm_b, w.m0, 3, 32, TOK*32);
  gemm(st, w, w.m0, bw.c1_w, bw.c1_b, nullptr, w.m1, 32, 32, 1, 0, 0, 0, 0);
  gemm(st, w, w.m1, bw.c2_w, bw.c2_b, nullptr, w.t0b, 32, 32, 1, 0, 0, 0, 0);
  k_dwconv<<<CDIV(TOK*32,256),256,0,st>>>(w.t0b, bw.dw_w, bw.dw_b, w.t1b, 32, 5, 2, 0, TOK*32);
  k_combine<<<CDIV(TOK*32,256),256,0,st>>>(w.m1, w.t1b, w.mf, TOK*32);
  // attention
  k_attn<<<BATCH*heads,256,0,st>>>(w.qkvb, ld, hd, dh, heads, bw.rescale, w.attnw);
  k_apply<<<TOK/256,256,0,st>>>(w.qkvb, ld, hd, dh, heads, w.mf, w.attnw, w.vmask, w.outc);
  gemm(st, w, w.outc, bw.proj_w, bw.proj_b, nullptr, w.proj, 32, 32, 0, 0, 0, 0, 0);
  // positional branch on masked v
  k_dwconv<<<CDIV(TOK*32,256),256,0,st>>>(w.vmask, bw.pos1, nullptr, w.p1, 32, 3, 1, 1, TOK*32);
  k_dwconv<<<CDIV(TOK*32,256),256,0,st>>>(w.p1, bw.pos2, nullptr, w.p2, 32, 3, 1, 0, TOK*32);
  k_add3<<<CDIV(TOK*32,256),256,0,st>>>(zout, w.proj, w.p2, zin, TOK*32);
  // FFN
  k_ln<<<CDIV(TOK,256),256,0,st>>>(zout, bw.ln_g, bw.ln_b, w.lnb, 32, TOK);
  gemm(st, w, w.lnb, bw.ff1, nullptr, nullptr, w.f1, 128, 32, 1, 1, 0, 0, 0);
  k_dwconv<<<CDIV(TOK*128,256),256,0,st>>>(w.f1, bw.ffd, nullptr, w.f2, 128, 3, 1, 1, TOK*128);
  gemm(st, w, w.f2, bw.ff2, nullptr, zout, zout, 32, 128, 1, 0, 0, 0, 0);
}

extern "C" void kernel_launch(void* const* d_in, const int* in_sizes, int n_in,
                              void* d_out, int out_size, void* d_ws, size_t ws_size,
                              hipStream_t stream)
{
  (void)in_sizes; (void)n_in; (void)out_size; (void)ws_size;
  const float* x = (const float*)d_in[0];
  float* out = (float*)d_out;
  auto P = [&](int i){ return (const float*)d_in[i]; };
  // jax sorted-key flatten: x=0; down=1..38; eproj=39..52; experts=53..318;
  // gce_b1=319 gce_b2=320 gce_w1=321 gce_w2=322; init_b=323 init_w=324;
  // oproj_b=325 oproj_w=326; orch=327..364; xi_blk=365..372 (cg1_b,cg1_w,cg2_b,
  // cg2_w,lc_b,lc_w,ln_b,ln_g); xi_c1_b=373 xi_c1_w=374 xi_c2_b=375 xi_c2_w=376
  // xi_ln_b=377 xi_ln_g=378

  char* cur = (char*)d_ws;
  auto alloc = [&](size_t elems) {
    float* p = (float*)cur;
    cur += ((elems * sizeof(float) + 255) & ~(size_t)255);
    return p;
  };
  WS w;
  w.vpool = alloc(16);
  w.ctx   = alloc(BATCH * 256);
  w.feat  = alloc((size_t)TOK * 32);
  w.zA    = alloc((size_t)TOK * 32);
  w.zB    = alloc((size_t)TOK * 32);
  w.zC    = alloc((size_t)TOK * 32);
  w.xq    = alloc((size_t)TOK * 64);
  w.qkvb  = alloc((size_t)TOK * 96);
  w.m0    = alloc((size_t)TOK * 32);
  w.m1    = alloc((size_t)TOK * 32);
  w.t0b   = alloc((size_t)TOK * 32);
  w.t1b   = alloc((size_t)TOK * 32);
  w.mf    = alloc((size_t)TOK * 32);
  w.attnw = alloc((size_t)BATCH * 2 * 1024);
  w.vmask = alloc((size_t)TOK * 32);
  w.outc  = alloc((size_t)TOK * 32);
  w.proj  = alloc((size_t)TOK * 32);
  w.p1    = alloc((size_t)TOK * 32);
  w.p2    = alloc((size_t)TOK * 32);
  w.lnb   = alloc((size_t)TOK * 32);
  w.f1    = alloc((size_t)TOK * 128);
  w.f2    = alloc((size_t)TOK * 128);
  w.z16   = alloc((size_t)TOK * 16);
  w.zn16  = alloc((size_t)TOK * 16);
  w.g16   = alloc(64);
  w.gate  = alloc(64);
  w.lc    = alloc((size_t)TOK * 16);
  w.comb  = alloc((size_t)TOK * 16);
  w.t3    = alloc((size_t)TOK * 3);
  w.wB    = (u16*)alloc(8192);   // bf16 weight staging [Npad,K], max 16K elems

  // GCE context vectors
  k_pool_nchw<<<BATCH*3,256,0,stream>>>(x, w.vpool);
  k_gce<<<1,256,0,stream>>>(w.vpool, P(321), P(319), P(322), P(320), w.ctx);
  // init conv 3->32 (3x3, pad 1)
  k_conv3x3<<<CDIV(TOK*32,256),256,0,stream>>>(x, P(324), P(323), w.feat, 3, 32, TOK*32);

  // shared down path (2 blocks, heads=2, dh=16, no ctx)
  BlkW d0 = blk_at(d_in, 1), d1 = blk_at(d_in, 20);
  msab_block(stream, w, x, w.feat, w.zA, d0, nullptr, 0, 2, 16);
  msab_block(stream, w, x, w.zA, w.zB, d1, nullptr, 0, 2, 16);  // zB = shared features

  // 7 experts (heads=2, dh=16, ctx g=i) -> d_out channels [i*9, i*9+9)
  for (int i = 0; i < 7; ++i) {
    BlkW e0 = blk_at(d_in, 53 + i*38), e1 = blk_at(d_in, 53 + i*38 + 19);
    msab_block(stream, w, x, w.zB, w.zA, e0, w.ctx, i, 2, 16);
    msab_block(stream, w, x, w.zA, w.zC, e1, w.ctx, i, 2, 16);
    gemm(stream, w, w.zC, P(40+2*i), P(39+2*i), nullptr, out, 9, 32, 1, 0, 1, 108, i*9);
  }

  // orchestrator (heads=1, dh=32, ctx g=7) -> channels [63,105)
  BlkW o0 = blk_at(d_in, 327), o1 = blk_at(d_in, 346);
  msab_block(stream, w, x, w.zB, w.zA, o0, w.ctx, 7, 1, 32);
  msab_block(stream, w, x, w.zA, w.zC, o1, w.ctx, 7, 1, 32);
  gemm(stream, w, w.zC, P(326), P(325), nullptr, out, 42, 32, 1, 0, 1, 108, 63);

  // xi net -> channels [105,108)
  k_conv3x3<<<CDIV(TOK*16,256),256,0,stream>>>(x, P(374), P(373), w.z16, 3, 16, TOK*16);
  k_ln<<<CDIV(TOK,256),256,0,stream>>>(w.z16, P(372), P(371), w.zn16, 16, TOK);
  k_pool_tok<<<BATCH*16,256,0,stream>>>(w.zn16, w.g16, 16);
  k_xi_gate<<<BATCH,32,0,stream>>>(w.g16, P(366), P(365), P(368), P(367), w.gate);
  k_dwconv<<<CDIV(TOK*16,256),256,0,stream>>>(w.zn16, P(370), P(369), w.lc, 16, 3, 1, 0, TOK*16);
  k_xi_combine<<<CDIV(TOK*16,256),256,0,stream>>>(w.lc, w.gate, w.z16, w.comb, TOK*16);
  k_pw_tok<<<CDIV(TOK*3,256),256,0,stream>>>(w.comb, P(376), P(375), w.t3, 16, 3, TOK*3);
  k_ln_nchw<<<CDIV(TOK,256),256,0,stream>>>(w.t3, P(378), P(377), out, 3, 108, 105);
}
